// Attention_11338713661523
// MI455X (gfx1250) — compile-verified, matrix operand order checked by reading, not run on hardware
//
#include <hip/hip_runtime.h>
#include <hip/hip_bf16.h>
#include <stdint.h>

// ---------------------------------------------------------------------------
// CDNA5 (gfx1250) wave32 WMMA + async-LDS implementation of the SVDQ
// joint-attention block. bf16 operands, fp32 accumulate.
// ---------------------------------------------------------------------------

typedef __attribute__((ext_vector_type(16))) __bf16 v16bf;
typedef __attribute__((ext_vector_type(8)))  float  v8f;

#define BM 128
#define BN 128
#define BK 32
#define LDT 40   // padded K-stride in LDS (elements): 80B rows -> conflict-free b128 reads

// CDNA5 async global->LDS copy (16B per lane), tracked by ASYNCcnt.
__device__ __forceinline__ void async_copy_b128(const void* gsrc, __bf16* lds_dst) {
  uint32_t l = (uint32_t)(uintptr_t)lds_dst;  // LDS aperture: low 32 bits = LDS offset
  asm volatile("global_load_async_to_lds_b128 %0, %1, off"
               :: "v"(l), "v"(gsrc) : "memory");
}
__device__ __forceinline__ void wait_async0() {
  asm volatile("s_wait_asynccnt 0x0" ::: "memory");
}

// Load one 16x32 bf16 WMMA fragment row for this lane.
// Elements 0..7  -> K = kg*8 + i ; elements 8..15 -> K = 16 + kg*8 + (i-8)
__device__ __forceinline__ v16bf frag_from_row(const __bf16* p, int kg) {
  union { uint4 u[2]; v16bf v; } r;
  r.u[0] = *(const uint4*)(p + kg * 8);
  r.u[1] = *(const uint4*)(p + 16 + kg * 8);
  return r.v;
}

template <typename T> struct TileLoad;
// fp32 source: convert in-register, store to LDS (sync; used only for softmax probs).
template <> struct TileLoad<float> {
  static __device__ __forceinline__ void go(const float* __restrict__ src, __bf16* dst) {
    const float4* p = (const float4*)src;
#pragma unroll
    for (int j = 0; j < 4; ++j) {
      float4 v = p[j];
      dst[4 * j + 0] = (__bf16)v.x;
      dst[4 * j + 1] = (__bf16)v.y;
      dst[4 * j + 2] = (__bf16)v.z;
      dst[4 * j + 3] = (__bf16)v.w;
    }
  }
};
// bf16 source: async DMA straight into LDS, no VGPR round trip.
template <> struct TileLoad<__bf16> {
  static __device__ __forceinline__ void go(const __bf16* __restrict__ src, __bf16* dst) {
    async_copy_b128(src, dst);
    async_copy_b128(src + 8, dst + 8);
  }
};

// C[M,N](f32) = A[M,K] x B[N,K]^T (+ rank-32 term t[M,32] x upT[N,32]^T) (+ bias[N])
// Double-buffered LDS; bf16 operands staged via global_load_async_to_lds_b128.
template <typename TA, typename TB>
__global__ __launch_bounds__(256) void gemm_wmma_kernel(
    const TA* __restrict__ A, int lda, size_t sAz,
    const TB* __restrict__ B, int ldb, size_t sBz,
    float* __restrict__ C, int ldc, size_t sCz,
    int K,
    const __bf16* __restrict__ tlr,
    const __bf16* __restrict__ upT,
    const float* __restrict__ bias) {
  __shared__ __bf16 At[2][BM * LDT];
  __shared__ __bf16 Bt[2][BN * LDT];

  A += (size_t)blockIdx.z * sAz;
  B += (size_t)blockIdx.z * sBz;
  C += (size_t)blockIdx.z * sCz;

  const int tid  = threadIdx.x;
  const int lane = tid & 31;
  const int wid  = tid >> 5;        // 8 waves
  const int wm0  = (wid & 1) * 64;  // wave tile: 64 rows x 32 cols
  const int wn0  = (wid >> 1) * 32;
  const int m0   = blockIdx.y * BM;
  const int n0   = blockIdx.x * BN;
  const int lrow = lane & 15;
  const int kg   = lane >> 4;

  const int loadRow = tid >> 1;        // 0..127
  const int loadCol = (tid & 1) * 16;  // 0 or 16

  v8f acc[4][2];
  const v8f vzero = {0.f, 0.f, 0.f, 0.f, 0.f, 0.f, 0.f, 0.f};
#pragma unroll
  for (int i = 0; i < 4; ++i)
#pragma unroll
    for (int j = 0; j < 2; ++j) acc[i][j] = vzero;

  // SVDQ low-rank term: exactly one K=32 WMMA per output fragment.
  if (tlr != nullptr) {
    v16bf tf[4], uf[2];
#pragma unroll
    for (int i = 0; i < 4; ++i)
      tf[i] = frag_from_row(tlr + (size_t)(m0 + wm0 + i * 16 + lrow) * 32, kg);
#pragma unroll
    for (int j = 0; j < 2; ++j)
      uf[j] = frag_from_row(upT + (size_t)(n0 + wn0 + j * 16 + lrow) * 32, kg);
#pragma unroll
    for (int i = 0; i < 4; ++i)
#pragma unroll
      for (int j = 0; j < 2; ++j)
        acc[i][j] = __builtin_amdgcn_wmma_f32_16x16x32_bf16(
            false, tf[i], false, uf[j], (short)0, acc[i][j], false, false);
  }

  const TA* aRow = A + (size_t)(m0 + loadRow) * lda + loadCol;
  const TB* bRow = B + (size_t)(n0 + loadRow) * ldb + loadCol;
  const int ldsOff = loadRow * LDT + loadCol;

  // prologue: stage tile 0 into buffer 0
  TileLoad<TA>::go(aRow, At[0] + ldsOff);
  TileLoad<TB>::go(bRow, Bt[0] + ldsOff);

  const int nIter = K / BK;
  for (int it = 0; it < nIter; ++it) {
    const int cur = it & 1;
    wait_async0();      // my async fills of buffer `cur` are done
    __syncthreads();    // everyone's fills of `cur` done; reads of `cur^1` done
    if (it + 1 < nIter) {
      const int k1 = (it + 1) * BK;
      TileLoad<TA>::go(aRow + k1, At[cur ^ 1] + ldsOff);
      TileLoad<TB>::go(bRow + k1, Bt[cur ^ 1] + ldsOff);
    }

    v16bf af[4], bfr[2];
#pragma unroll
    for (int i = 0; i < 4; ++i)
      af[i] = frag_from_row(At[cur] + (wm0 + i * 16 + lrow) * LDT, kg);
#pragma unroll
    for (int j = 0; j < 2; ++j)
      bfr[j] = frag_from_row(Bt[cur] + (wn0 + j * 16 + lrow) * LDT, kg);
#pragma unroll
    for (int i = 0; i < 4; ++i)
#pragma unroll
      for (int j = 0; j < 2; ++j)
        acc[i][j] = __builtin_amdgcn_wmma_f32_16x16x32_bf16(
            false, af[i], false, bfr[j], (short)0, acc[i][j], false, false);
  }

#pragma unroll
  for (int i = 0; i < 4; ++i)
#pragma unroll
    for (int j = 0; j < 2; ++j) {
      const int col  = n0 + wn0 + j * 16 + lrow;
      const float bb = bias ? bias[col] : 0.f;
#pragma unroll
      for (int r = 0; r < 8; ++r) {
        const int row = m0 + wm0 + i * 16 + kg * 8 + r;  // C/D lane layout
        C[(size_t)row * ldc + col] = acc[i][j][r] + bb;
      }
    }
}

// dst[i] = (bf16)src[i]; n must be a multiple of 2048 (8 elems / thread).
__global__ __launch_bounds__(256) void cvt_bf16_kernel(
    const float* __restrict__ src, __bf16* __restrict__ dst, size_t n) {
  const size_t i = ((size_t)blockIdx.x * 256 + threadIdx.x) * 8;
  if (i + 8 > n) return;
  float4 a = *(const float4*)(src + i);
  float4 b = *(const float4*)(src + i + 4);
  __bf16 o[8];
  o[0] = (__bf16)a.x; o[1] = (__bf16)a.y; o[2] = (__bf16)a.z; o[3] = (__bf16)a.w;
  o[4] = (__bf16)b.x; o[5] = (__bf16)b.y; o[6] = (__bf16)b.z; o[7] = (__bf16)b.w;
  *(uint4*)(dst + i) = *(const uint4*)o;
}

// t[m, 0..31] = sum_k A[m,k] * down[k, 0..31]   (bf16 output, one block per row)
__global__ __launch_bounds__(256) void lowrank_t_kernel(
    const float* __restrict__ A, int lda, const float* __restrict__ down,
    __bf16* __restrict__ t, int K) {
  const int m   = blockIdx.x;
  const int tid = threadIdx.x;
  const int r    = tid & 31;
  const int part = tid >> 5;
  const float* a = A + (size_t)m * lda;
  float acc = 0.f;
  for (int k = part; k < K; k += 8) acc += a[k] * down[(size_t)k * 32 + r];
  __shared__ float red[256];
  red[tid] = acc;
  __syncthreads();
  if (tid < 32) {
    float s = 0.f;
#pragma unroll
    for (int p = 0; p < 8; ++p) s += red[r + 32 * p];
    t[(size_t)m * 32 + r] = (__bf16)s;
  }
}

// upT[n, r] = up[r, n]  (f32 [32,N] -> bf16 [N,32])
__global__ __launch_bounds__(256) void transpose_up_kernel(
    const float* __restrict__ up, __bf16* __restrict__ upT, int N) {
  const int n = blockIdx.x * blockDim.x + threadIdx.x;
  if (n >= N) return;
#pragma unroll
  for (int r = 0; r < 32; ++r)
    upT[(size_t)n * 32 + r] = (__bf16)up[(size_t)r * N + n];
}

// Fused: concat(txt,img) -> split q/k/v -> RMS-norm q,k -> RoPE q,k -> repack
// q,k as bf16 [H][S][HD]; v transposed as bf16 [H][HD][S].
__global__ __launch_bounds__(128) void qkv_post_kernel(
    const float* __restrict__ qkv_txt, const float* __restrict__ qkv_img,
    const float* __restrict__ nqw, const float* __restrict__ nkw,
    const float* __restrict__ naqw, const float* __restrict__ nakw,
    const float* __restrict__ cosP, const float* __restrict__ sinP,
    __bf16* __restrict__ qh, __bf16* __restrict__ kh, __bf16* __restrict__ vt) {
  const int S = 2048, ST = 512, N3 = 9216, HD = 128;
  const int s = blockIdx.x, h = blockIdx.y, d = threadIdx.x;

  const float* row;
  const float *wq, *wk;
  if (s < ST) { row = qkv_txt + (size_t)s * N3;        wq = naqw; wk = nakw; }
  else        { row = qkv_img + (size_t)(s - ST) * N3; wq = nqw;  wk = nkw;  }

  const float qv = row[h * HD + d];
  const float kv = row[3072 + h * HD + d];
  const float vv = row[6144 + h * HD + d];

  __shared__ float red[128];
  __shared__ float bq[128];
  __shared__ float bk[128];

  red[d] = qv * qv;
  __syncthreads();
  for (int off = 64; off; off >>= 1) {
    if (d < off) red[d] += red[d + off];
    __syncthreads();
  }
  const float invq = rsqrtf(red[0] * (1.f / 128.f) + 1e-5f);
  __syncthreads();
  red[d] = kv * kv;
  __syncthreads();
  for (int off = 64; off; off >>= 1) {
    if (d < off) red[d] += red[d + off];
    __syncthreads();
  }
  const float invk = rsqrtf(red[0] * (1.f / 128.f) + 1e-5f);

  bq[d] = qv * invq * wq[d];
  bk[d] = kv * invk * wk[d];
  __syncthreads();

  if (d < 64) {
    const float c  = cosP[(size_t)s * 64 + d];
    const float sn = sinP[(size_t)s * 64 + d];
    const float q0 = bq[2 * d], q1 = bq[2 * d + 1];
    const float k0 = bk[2 * d], k1 = bk[2 * d + 1];
    const size_t base = ((size_t)h * S + s) * HD;
    qh[base + 2 * d]     = (__bf16)(q0 * c - q1 * sn);
    qh[base + 2 * d + 1] = (__bf16)(q0 * sn + q1 * c);
    kh[base + 2 * d]     = (__bf16)(k0 * c - k1 * sn);
    kh[base + 2 * d + 1] = (__bf16)(k0 * sn + k1 * c);
  }
  vt[((size_t)h * HD + d) * S + s] = (__bf16)vv;
}

// In-place scaled row softmax; one block per row.
__global__ __launch_bounds__(256) void softmax_kernel(float* __restrict__ s,
                                                      int cols, float scale) {
  float* row = s + (size_t)blockIdx.x * cols;
  __shared__ float red[256];
  const int t = threadIdx.x;
  float m = -1e30f;
  for (int c = t; c < cols; c += 256) m = fmaxf(m, row[c] * scale);
  red[t] = m;
  __syncthreads();
  for (int off = 128; off; off >>= 1) {
    if (t < off) red[t] = fmaxf(red[t], red[t + off]);
    __syncthreads();
  }
  const float mx = red[0];
  __syncthreads();
  float sum = 0.f;
  for (int c = t; c < cols; c += 256) {
    const float e = __expf(row[c] * scale - mx);
    row[c] = e;
    sum += e;
  }
  red[t] = sum;
  __syncthreads();
  for (int off = 128; off; off >>= 1) {
    if (t < off) red[t] += red[t + off];
    __syncthreads();
  }
  const float inv = 1.f / red[0];
  __syncthreads();
  for (int c = t; c < cols; c += 256) row[c] *= inv;
}

extern "C" void kernel_launch(void* const* d_in, const int* in_sizes, int n_in,
                              void* d_out, int out_size, void* d_ws, size_t ws_size,
                              hipStream_t stream) {
  (void)in_sizes; (void)n_in; (void)out_size; (void)ws_size;
  const float* x_img     = (const float*)d_in[0];   // [1536,3072]
  const float* x_txt     = (const float*)d_in[1];   // [512,3072]
  const float* rope_cos  = (const float*)d_in[2];   // [2048,64]
  const float* rope_sin  = (const float*)d_in[3];
  const float* qkv_w     = (const float*)d_in[4];   // [9216,3072]
  const float* qkv_down  = (const float*)d_in[5];   // [3072,32]
  const float* qkv_up    = (const float*)d_in[6];   // [32,9216]
  const float* aqkv_w    = (const float*)d_in[7];
  const float* aqkv_down = (const float*)d_in[8];
  const float* aqkv_up   = (const float*)d_in[9];
  const float* out_w     = (const float*)d_in[10];  // [3072,3072]
  const float* out_down  = (const float*)d_in[11];  // [3072,32]
  const float* out_up    = (const float*)d_in[12];  // [32,3072]
  const float* out_b     = (const float*)d_in[13];
  const float* aout_w    = (const float*)d_in[14];
  const float* aout_down = (const float*)d_in[15];
  const float* aout_up   = (const float*)d_in[16];
  const float* aout_b    = (const float*)d_in[17];
  const float* nqw       = (const float*)d_in[18];
  const float* nkw       = (const float*)d_in[19];
  const float* naqw      = (const float*)d_in[20];
  const float* nakw      = (const float*)d_in[21];
  float* outp = (float*)d_out;

  const int S = 2048, ST = 512, SI = 1536, DD = 3072, HH = 24, HD = 128, N3 = 9216;
  const int G = 4;  // heads per attention batch group
  const float scale = 0.08838834764831845f;  // 1/sqrt(128)

  char* ws = (char*)d_ws;
  size_t off = 0;
  auto alloc = [&](size_t bytes) -> char* {
    char* p = ws + off;
    off = (off + bytes + 255) & ~(size_t)255;
    return p;
  };
  float*  qkv_img  = (float*)alloc((size_t)SI * N3 * 4);   // dead after qkv_post
  float*  qkv_txt  = (float*)alloc((size_t)ST * N3 * 4);   // dead after qkv_post
  __bf16* qh       = (__bf16*)alloc((size_t)HH * S * HD * 2);
  __bf16* kh       = (__bf16*)alloc((size_t)HH * S * HD * 2);
  __bf16* vt       = (__bf16*)alloc((size_t)HH * HD * S * 2);
  float*  attn     = (float*)alloc((size_t)S * DD * 4);
  __bf16* attn_b   = (__bf16*)alloc((size_t)S * DD * 2);
  __bf16* xb_img   = (__bf16*)alloc((size_t)SI * DD * 2);
  __bf16* xb_txt   = (__bf16*)alloc((size_t)ST * DD * 2);
  __bf16* wb_qkv   = (__bf16*)alloc((size_t)N3 * DD * 2);
  __bf16* wb_aqkv  = (__bf16*)alloc((size_t)N3 * DD * 2);
  __bf16* wb_out   = (__bf16*)alloc((size_t)DD * DD * 2);
  __bf16* wb_aout  = (__bf16*)alloc((size_t)DD * DD * 2);
  __bf16* t_img_q  = (__bf16*)alloc((size_t)SI * 32 * 2);
  __bf16* t_txt_q  = (__bf16*)alloc((size_t)ST * 32 * 2);
  __bf16* t_img_o  = (__bf16*)alloc((size_t)SI * 32 * 2);
  __bf16* t_txt_o  = (__bf16*)alloc((size_t)ST * 32 * 2);
  __bf16* upT_qkv  = (__bf16*)alloc((size_t)N3 * 32 * 2);
  __bf16* upT_aqkv = (__bf16*)alloc((size_t)N3 * 32 * 2);
  __bf16* upT_out  = (__bf16*)alloc((size_t)DD * 32 * 2);
  __bf16* upT_aout = (__bf16*)alloc((size_t)DD * 32 * 2);
  // Scores (G*2048*2048 f32 = 67 MB) overlays the dead qkv scratch (75 MB).
  float* scores = qkv_img;

  auto cvt = [&](const float* s_, __bf16* d_, size_t n) {
    cvt_bf16_kernel<<<dim3((unsigned)(n / 2048)), 256, 0, stream>>>(s_, d_, n);
  };

  // --- bf16 operand prep (enables async-LDS streaming in all GEMMs) --------
  cvt(qkv_w,  wb_qkv,  (size_t)N3 * DD);
  cvt(aqkv_w, wb_aqkv, (size_t)N3 * DD);
  cvt(out_w,  wb_out,  (size_t)DD * DD);
  cvt(aout_w, wb_aout, (size_t)DD * DD);
  cvt(x_img,  xb_img,  (size_t)SI * DD);
  cvt(x_txt,  xb_txt,  (size_t)ST * DD);

  // --- SVDQ prep -----------------------------------------------------------
  transpose_up_kernel<<<dim3(N3 / 256), 256, 0, stream>>>(qkv_up, upT_qkv, N3);
  transpose_up_kernel<<<dim3(N3 / 256), 256, 0, stream>>>(aqkv_up, upT_aqkv, N3);
  transpose_up_kernel<<<dim3(DD / 256), 256, 0, stream>>>(out_up, upT_out, DD);
  transpose_up_kernel<<<dim3(DD / 256), 256, 0, stream>>>(aout_up, upT_aout, DD);
  lowrank_t_kernel<<<dim3(SI), 256, 0, stream>>>(x_img, DD, qkv_down, t_img_q, DD);
  lowrank_t_kernel<<<dim3(ST), 256, 0, stream>>>(x_txt, DD, aqkv_down, t_txt_q, DD);

  // --- QKV projections (x @ W.T + t @ upT.T), all-async bf16 ---------------
  gemm_wmma_kernel<__bf16, __bf16><<<dim3(N3 / BN, SI / BM, 1), 256, 0, stream>>>(
      xb_img, DD, 0, wb_qkv, DD, 0, qkv_img, N3, 0, DD, t_img_q, upT_qkv, nullptr);
  gemm_wmma_kernel<__bf16, __bf16><<<dim3(N3 / BN, ST / BM, 1), 256, 0, stream>>>(
      xb_txt, DD, 0, wb_aqkv, DD, 0, qkv_txt, N3, 0, DD, t_txt_q, upT_aqkv, nullptr);

  // --- RMS-norm + RoPE + repack --------------------------------------------
  qkv_post_kernel<<<dim3(S, HH), 128, 0, stream>>>(
      qkv_txt, qkv_img, nqw, nkw, naqw, nakw, rope_cos, rope_sin, qh, kh, vt);

  // --- Attention: groups of G heads batched over gridDim.z -----------------
  for (int g = 0; g < HH; g += G) {
    gemm_wmma_kernel<__bf16, __bf16><<<dim3(S / BN, S / BM, G), 256, 0, stream>>>(
        qh + (size_t)g * S * HD, HD, (size_t)S * HD,
        kh + (size_t)g * S * HD, HD, (size_t)S * HD,
        scores, S, (size_t)S * S, HD, nullptr, nullptr, nullptr);
    softmax_kernel<<<dim3(G * S), 256, 0, stream>>>(scores, S, scale);
    gemm_wmma_kernel<float, __bf16><<<dim3(HD / BN, S / BM, G), 256, 0, stream>>>(
        scores, S, (size_t)S * S,
        vt + (size_t)g * HD * S, S, (size_t)HD * S,
        attn + (size_t)g * HD, DD, (size_t)HD, S, nullptr, nullptr, nullptr);
  }

  // --- Output projections ---------------------------------------------------
  cvt(attn, attn_b, (size_t)S * DD);
  const float* img_attn = attn + (size_t)ST * DD;
  lowrank_t_kernel<<<dim3(SI), 256, 0, stream>>>(img_attn, DD, out_down, t_img_o, DD);
  lowrank_t_kernel<<<dim3(ST), 256, 0, stream>>>(attn, DD, aout_down, t_txt_o, DD);

  gemm_wmma_kernel<__bf16, __bf16><<<dim3(DD / BN, SI / BM, 1), 256, 0, stream>>>(
      attn_b + (size_t)ST * DD, DD, 0, wb_out, DD, 0, outp, DD, 0, DD,
      t_img_o, upT_out, out_b);
  gemm_wmma_kernel<__bf16, __bf16><<<dim3(DD / BN, ST / BM, 1), 256, 0, stream>>>(
      attn_b, DD, 0, wb_aout, DD, 0, outp + (size_t)SI * DD, DD, 0, DD,
      t_txt_o, upT_aout, aout_b);
}
